// SPLICE_75849122447417
// MI455X (gfx1250) — matrix operand order repaired and checked by reading, not compiled, not run on hardware
//
#include <hip/hip_runtime.h>
#include <hip/hip_bf16.h>
#include <cmath>

// ---------------- problem constants ----------------
#define BATCH   128
#define DIM     1024
#define K_DICT  10000
#define K_PAD   10016            // 313 * 32
#define N_ITERS 200
#define POWER_ITERS 30
#define ALPHA_F 4.8828125e-6f    // 0.01 / (2*1024)

typedef __bf16 bf16_t;
typedef __bf16 v16bf __attribute__((ext_vector_type(16)));
typedef __bf16 v8bf  __attribute__((ext_vector_type(8)));
typedef float  v8f   __attribute__((ext_vector_type(8)));

// pointer types matching the async-copy builtin's signature:
//   void __builtin_amdgcn_global_load_async_to_lds_b128(v4i AS1*, v4i AS3*, imm, imm)
typedef int v4i_t __attribute__((vector_size(16)));
typedef __attribute__((address_space(1))) v4i_t* gvp_t;
typedef __attribute__((address_space(3))) v4i_t* lvp_t;

// ---------------- gfx1250 async global->LDS staging helpers ----------------
__device__ __forceinline__ void stage16(const bf16_t* g, bf16_t* l) {
#if __has_builtin(__builtin_amdgcn_global_load_async_to_lds_b128)
  __builtin_amdgcn_global_load_async_to_lds_b128((gvp_t)(bf16_t*)g, (lvp_t)l, 0, 0);
#else
  *(v8bf*)l = *(const v8bf*)g;   // fallback: global_load_b128 + ds_store_b128
#endif
}
__device__ __forceinline__ void stage_wait() {
#if __has_builtin(__builtin_amdgcn_s_wait_asynccnt)
  __builtin_amdgcn_s_wait_asynccnt(0);
#endif
}

// ---------------- one-time conversion: D -> bf16 (padded rows) and D^T -> bf16 (padded K) -------
__global__ void k_convert(const float* __restrict__ D, bf16_t* __restrict__ Db,
                          bf16_t* __restrict__ Dt) {
  long idx = (long)blockIdx.x * 256 + threadIdx.x;   // covers DIM * K_PAD exactly
  int d = (int)(idx / K_PAD);
  int k = (int)(idx % K_PAD);
  float val = (k < K_DICT) ? D[(long)k * DIM + d] : 0.0f;
  bf16_t b = (bf16_t)val;
  Dt[(long)d * K_PAD + k] = b;                       // D^T[d, k] = D[k, d]
  Db[(long)k * DIM + d]   = b;                       // D[k, d], rows 10000..10015 zero
}

// ---------------- init: zero z (f32), w (=d_out), z_bf16; v = ones; scalars = 0 ----------------
__global__ void k_init(float* z, float* w, bf16_t* zb, float* v, float* scal) {
  int idx = blockIdx.x * 256 + threadIdx.x;          // covers BATCH*K_PAD = 1282048
  if (idx < BATCH * K_DICT) { z[idx] = 0.0f; w[idx] = 0.0f; }
  if (idx < BATCH * K_PAD)  { zb[idx] = (bf16_t)0.0f; }
  if (idx < K_DICT)         { v[idx] = 1.0f; }
  if (idx < 8)              { scal[idx] = 0.0f; }
}

// ---------------- power iteration: u_part[b] = partial (v @ D) over a k-chunk ----------------
__global__ void k_matvec_u(const float* __restrict__ v, const float* __restrict__ D,
                           float* __restrict__ u_part) {
  int b = blockIdx.x, t = threadIdx.x;               // 40 blocks, 250 rows each
  float a0 = 0.f, a1 = 0.f, a2 = 0.f, a3 = 0.f;
  int k0 = b * 250, k1 = k0 + 250;
  for (int k = k0; k < k1; ++k) {
    float s = v[k];
    const float* row = D + (long)k * DIM;
    a0 += s * row[t];       a1 += s * row[256 + t];
    a2 += s * row[512 + t]; a3 += s * row[768 + t];
  }
  float* up = u_part + (long)b * DIM;
  up[t] = a0; up[256 + t] = a1; up[512 + t] = a2; up[768 + t] = a3;
}

// deterministic fixed-order reduce of the 40 partials
__global__ void k_reduce_u(const float* __restrict__ u_part, float* __restrict__ u) {
  int col = blockIdx.x * 256 + threadIdx.x;          // 4 blocks -> 1024 cols
  float a = 0.f;
  for (int b = 0; b < 40; ++b) a += u_part[(long)b * DIM + col];
  u[col] = a;
}

// v2[k] = (u . D[k,:]) / 1024     (one block per dictionary row, LDS tree reduce)
__global__ void k_matvec_v(const float* __restrict__ u, const float* __restrict__ D,
                           float* __restrict__ v2) {
  __shared__ float red[256];
  int k = blockIdx.x, t = threadIdx.x;
  const float* row = D + (long)k * DIM;
  float a = u[t] * row[t] + u[256 + t] * row[256 + t] +
            u[512 + t] * row[512 + t] + u[768 + t] * row[768 + t];
  red[t] = a; __syncthreads();
  for (int s = 128; s > 0; s >>= 1) { if (t < s) red[t] += red[t + s]; __syncthreads(); }
  if (t == 0) v2[k] = red[0] * (1.0f / 1024.0f);
}

// sum v2^2 -> inv_norm, L, step, thresh   (single block; deterministic)
__global__ void k_vreduce(const float* __restrict__ v2, float* __restrict__ scal) {
  __shared__ float red[256];
  int t = threadIdx.x;
  float a = 0.f;
  for (int k = t; k < K_DICT; k += 256) { float x = v2[k]; a += x * x; }
  red[t] = a; __syncthreads();
  for (int s = 128; s > 0; s >>= 1) { if (t < s) red[t] += red[t + s]; __syncthreads(); }
  if (t == 0) {
    float L = sqrtf(red[0]);
    scal[0] = 1.0f / (L + 1e-12f);   // inv norm of v2
    scal[1] = L;
    float step = 1.0f / L;
    scal[2] = step;
    scal[3] = step * ALPHA_F;
  }
}

__global__ void k_normalize(const float* __restrict__ v2, const float* __restrict__ scal,
                            float* __restrict__ v) {
  int idx = blockIdx.x * 256 + threadIdx.x;
  if (idx < K_DICT) v[idx] = v2[idx] * scal[0];
}

// ---------------- GEMM1: Rm_bf16 = bf16( z_bf16 @ D  -  Y )   M=128 N=1024 K=10016 ----------------
// block: 256 threads = 8 waves; wave w -> M tile w; block -> 32 N columns (2 tiles/wave).
// B tile (32K x 32N, 2 KB) staged in LDS via async global->LDS copies by waves 0-3.
__global__ void k_gemm1(const bf16_t* __restrict__ zb, const bf16_t* __restrict__ Dt,
                        const float* __restrict__ Y, bf16_t* __restrict__ Rm) {
  __shared__ __align__(128) bf16_t ldsB[32 * 32];    // [col][k] rows of 64B
  int tid = threadIdx.x;
  int lane = tid & 31, wid = tid >> 5;
  int lane15 = lane & 15, laneHalf = lane >> 4;
  int Mbase = wid * 16;
  int Nbase = blockIdx.x * 32;
  const bf16_t* arow = zb + (long)(Mbase + lane15) * K_PAD + laneHalf * 16;
  int scol = tid >> 2;                 // staging col (tid < 128)
  int schunk = (tid & 3) * 8;          // 16B chunk within 64B row
  const bf16_t* srow = Dt + (long)(Nbase + scol) * K_PAD + schunk;
  bf16_t* sdst = &ldsB[scol * 32 + schunk];
  v8f acc0 = {}, acc1 = {};
  for (int kk = 0; kk < K_PAD; kk += 32) {
    if (tid < 128) stage16(srow + kk, sdst);
    stage_wait();
    __syncthreads();
    v8bf a0 = *(const v8bf*)(arow + kk);
    v8bf a1 = *(const v8bf*)(arow + kk + 8);
    v16bf A;
#pragma unroll
    for (int e = 0; e < 8; ++e) { A[e] = a0[e]; A[8 + e] = a1[e]; }
    v16bf B0 = *(const v16bf*)&ldsB[lane15 * 32 + laneHalf * 16];
    v16bf B1 = *(const v16bf*)&ldsB[(16 + lane15) * 32 + laneHalf * 16];
    acc0 = __builtin_amdgcn_wmma_f32_16x16x32_bf16(false, A, false, B0, (short)0, acc0,
                                                   false, false);
    acc1 = __builtin_amdgcn_wmma_f32_16x16x32_bf16(false, A, false, B1, (short)0, acc1,
                                                   false, false);
    __syncthreads();
  }
#pragma unroll
  for (int r = 0; r < 8; ++r) {
    int m = Mbase + laneHalf * 8 + r;
    long idx0 = (long)m * DIM + Nbase + lane15;
    Rm[idx0]      = (bf16_t)(acc0[r] - Y[idx0]);
    Rm[idx0 + 16] = (bf16_t)(acc1[r] - Y[idx0 + 16]);
  }
}

// ---------------- GEMM2 + fused FISTA step ----------------
// G = Rm @ D^T ; g = G/1024 ; w_new = max(z - step*g - thresh, 0) ;
// z_new = w_new + c*(w_new - w_old) ; also refresh z_bf16.
// M=128 N=10016(pad) K=1024. 313 blocks x 256 threads, 32 N cols per block.
__global__ void k_gemm2_fista(const bf16_t* __restrict__ Rm, const bf16_t* __restrict__ Db,
                              const float* __restrict__ scal,
                              float* __restrict__ w, float* __restrict__ z,
                              bf16_t* __restrict__ zb, float coef) {
  __shared__ __align__(128) bf16_t ldsB[32 * 32];
  int tid = threadIdx.x;
  int lane = tid & 31, wid = tid >> 5;
  int lane15 = lane & 15, laneHalf = lane >> 4;
  int Mbase = wid * 16;
  int Nbase = blockIdx.x * 32;
  const bf16_t* arow = Rm + (long)(Mbase + lane15) * DIM + laneHalf * 16;
  int scol = tid >> 2;
  int schunk = (tid & 3) * 8;
  const bf16_t* srow = Db + (long)(Nbase + scol) * DIM + schunk;   // Db padded to K_PAD rows
  bf16_t* sdst = &ldsB[scol * 32 + schunk];
  v8f acc0 = {}, acc1 = {};
  for (int kk = 0; kk < DIM; kk += 32) {
    if (tid < 128) stage16(srow + kk, sdst);
    stage_wait();
    __syncthreads();
    v8bf a0 = *(const v8bf*)(arow + kk);
    v8bf a1 = *(const v8bf*)(arow + kk + 8);
    v16bf A;
#pragma unroll
    for (int e = 0; e < 8; ++e) { A[e] = a0[e]; A[8 + e] = a1[e]; }
    v16bf B0 = *(const v16bf*)&ldsB[lane15 * 32 + laneHalf * 16];
    v16bf B1 = *(const v16bf*)&ldsB[(16 + lane15) * 32 + laneHalf * 16];
    acc0 = __builtin_amdgcn_wmma_f32_16x16x32_bf16(false, A, false, B0, (short)0, acc0,
                                                   false, false);
    acc1 = __builtin_amdgcn_wmma_f32_16x16x32_bf16(false, A, false, B1, (short)0, acc1,
                                                   false, false);
    __syncthreads();
  }
  float step = scal[2], thresh = scal[3];
#pragma unroll
  for (int j = 0; j < 2; ++j) {
    int n = Nbase + j * 16 + lane15;
    if (n < K_DICT) {
      float g8[8] = {acc0[0], acc0[1], acc0[2], acc0[3], acc0[4], acc0[5], acc0[6], acc0[7]};
      if (j == 1) { g8[0]=acc1[0]; g8[1]=acc1[1]; g8[2]=acc1[2]; g8[3]=acc1[3];
                    g8[4]=acc1[4]; g8[5]=acc1[5]; g8[6]=acc1[6]; g8[7]=acc1[7]; }
#pragma unroll
      for (int r = 0; r < 8; ++r) {
        int m = Mbase + laneHalf * 8 + r;
        long idx = (long)m * K_DICT + n;
        float g = g8[r] * (1.0f / 1024.0f);
        float zo = z[idx], wo = w[idx];
        float wn = zo - step * g - thresh;
        wn = wn > 0.0f ? wn : 0.0f;
        float zn = wn + coef * (wn - wo);
        w[idx] = wn;
        z[idx] = zn;
        zb[(long)m * K_PAD + n] = (bf16_t)zn;
      }
    }
  }
}

// ---------------- recon = w @ D  (128 blocks, exploit sparsity of w, uniform branch) ----------------
__global__ void k_recon(const float* __restrict__ w, const float* __restrict__ D,
                        float* __restrict__ recon) {
  int m = blockIdx.x, t = threadIdx.x;
  float a0 = 0.f, a1 = 0.f, a2 = 0.f, a3 = 0.f;
  const float* wr = w + (long)m * K_DICT;
  for (int k = 0; k < K_DICT; ++k) {
    float s = wr[k];                    // uniform across the block
    if (s != 0.0f) {
      const float* row = D + (long)k * DIM;
      a0 += s * row[t];       a1 += s * row[256 + t];
      a2 += s * row[512 + t]; a3 += s * row[768 + t];
    }
  }
  float* rr = recon + (long)m * DIM;
  rr[t] = a0; rr[256 + t] = a1; rr[512 + t] = a2; rr[768 + t] = a3;
}

// row-normalize recon and dot with embedding -> per-row partial score
__global__ void k_rownorm_dot(const float* __restrict__ recon, const float* __restrict__ Y,
                              float* __restrict__ score_part) {
  __shared__ float red[256];
  __shared__ float snorm;
  int m = blockIdx.x, t = threadIdx.x;
  const float* r = recon + (long)m * DIM;
  const float* e = Y + (long)m * DIM;
  float ss = r[t] * r[t] + r[256 + t] * r[256 + t] +
             r[512 + t] * r[512 + t] + r[768 + t] * r[768 + t];
  red[t] = ss; __syncthreads();
  for (int s = 128; s > 0; s >>= 1) { if (t < s) red[t] += red[t + s]; __syncthreads(); }
  if (t == 0) snorm = 1.0f / (sqrtf(red[0]) + 1e-12f);
  __syncthreads();
  float inv = snorm;
  float d = inv * (r[t] * e[t] + r[256 + t] * e[256 + t] +
                   r[512 + t] * e[512 + t] + r[768 + t] * e[768 + t]);
  __syncthreads();
  red[t] = d; __syncthreads();
  for (int s = 128; s > 0; s >>= 1) { if (t < s) red[t] += red[t + s]; __syncthreads(); }
  if (t == 0) score_part[m] = red[0];
}

// deterministic final score reduce
__global__ void k_score(const float* __restrict__ score_part, float* __restrict__ score) {
  if (threadIdx.x == 0 && blockIdx.x == 0) {
    float s = 0.f;
    for (int m = 0; m < BATCH; ++m) s += score_part[m];
    score[0] = s;
  }
}

// ---------------- host side ----------------
extern "C" void kernel_launch(void* const* d_in, const int* in_sizes, int n_in,
                              void* d_out, int out_size, void* d_ws, size_t ws_size,
                              hipStream_t stream) {
  const float* Y = (const float*)d_in[0];   // embedding [128,1024]
  const float* D = (const float*)d_in[1];   // dictionary [10000,1024]

  float* out = (float*)d_out;
  float* w_out = out;                       // [128,10000] weights live here
  float* score = out + (long)BATCH * K_DICT;

  char* ws = (char*)d_ws;
  size_t off = 0;
  auto take = [&](size_t bytes) { char* p = ws + off; off += (bytes + 255) & ~(size_t)255; return p; };
  bf16_t* Db   = (bf16_t*)take((size_t)K_PAD * DIM * 2);      // 20.51 MB (padded rows)
  bf16_t* Dt   = (bf16_t*)take((size_t)DIM * K_PAD * 2);      // 20.51 MB
  bf16_t* zb   = (bf16_t*)take((size_t)BATCH * K_PAD * 2);    // 2.56 MB
  bf16_t* Rm   = (bf16_t*)take((size_t)BATCH * DIM * 2);      // 0.26 MB
  float*  z    = (float*) take((size_t)BATCH * K_DICT * 4);   // 5.12 MB
  float*  recon= (float*) take((size_t)BATCH * DIM * 4);      // 0.52 MB
  float*  u_part=(float*) take((size_t)40 * DIM * 4);
  float*  u    = (float*) take((size_t)DIM * 4);
  float*  v    = (float*) take((size_t)K_DICT * 4);
  float*  v2   = (float*) take((size_t)K_DICT * 4);
  float*  spart= (float*) take((size_t)BATCH * 4);
  float*  scal = (float*) take(64);

  // one-time bf16 conversion + transpose (D is L2-resident thereafter)
  k_convert<<<(DIM * K_PAD) / 256, 256, 0, stream>>>(D, Db, Dt);
  k_init<<<(BATCH * K_PAD) / 256, 256, 0, stream>>>(z, w_out, zb, v, scal);

  // Lipschitz power iteration (30 body iters + final norm -> L, step, thresh)
  for (int it = 0; it < POWER_ITERS + 1; ++it) {
    k_matvec_u<<<40, 256, 0, stream>>>(v, D, u_part);
    k_reduce_u<<<4, 256, 0, stream>>>(u_part, u);
    k_matvec_v<<<K_DICT, 256, 0, stream>>>(u, D, v2);
    k_vreduce<<<1, 256, 0, stream>>>(v2, scal);
    if (it < POWER_ITERS) k_normalize<<<40, 256, 0, stream>>>(v2, scal, v);
  }

  // FISTA main loop: momentum coefficients are data-independent -> precompute on host
  double t = 1.0;
  for (int i = 0; i < N_ITERS; ++i) {
    double tn = 0.5 * (1.0 + sqrt(1.0 + 4.0 * t * t));
    float c = (float)((t - 1.0) / tn);
    t = tn;
    k_gemm1<<<DIM / 32, 256, 0, stream>>>(zb, Dt, Y, Rm);
    k_gemm2_fista<<<K_PAD / 32, 256, 0, stream>>>(Rm, Db, scal, w_out, z, zb, c);
  }

  // recompose + score
  k_recon<<<BATCH, 256, 0, stream>>>(w_out, D, recon);
  k_rownorm_dot<<<BATCH, 256, 0, stream>>>(recon, Y, spart);
  k_score<<<1, 64, 0, stream>>>(spart, score);

  (void)in_sizes; (void)n_in; (void)out_size; (void)ws_size;
}
